// QuantizedLinear_833223655649
// MI455X (gfx1250) — compile-verified, matrix operand order checked by reading, not run on hardware
//
#include <hip/hip_runtime.h>

// ---------------------------------------------------------------------------
// Quantized linear: Y[M,N] = (X[M,K] · (Wq[N,K] - zp)^T) * scale + bias
// M = 8192, K = 4096, N = 16384.  Compute-bound -> bf16 WMMA.
// - W staged as bf16(q - zp): exact (|q|<=128 fits bf16 significand);
//   scale applied in fp32 epilogue.
// - X staged as bf16 with 1 add (round bias) + v_perm pack per pair.
// - 128x128 block tile, 8 waves (4x2), 32x64 wave tile = 8 WMMA / K-step.
// - L2 chunk swizzle keeps the streamed W window (32 MB) resident in the
//   192 MB L2 so HBM traffic (~1.9 GB) stays far under the compute time.
// ---------------------------------------------------------------------------

typedef __attribute__((ext_vector_type(16))) __bf16 v16bf;
typedef __attribute__((ext_vector_type(8)))  float  v8f;

#define M_DIM 8192
#define N_DIM 16384
#define K_DIM 4096

#define BM 128
#define BN 128
#define BK 32
#define NKB (K_DIM / BK)          // 128
#define LDS_STRIDE 40             // 32 bf16 + pad: conflict-free b128 frag reads
#define MBLKS (M_DIM / BM)        // 64
#define NBLKS (N_DIM / BN)        // 128
#define CHUNK_NBLKS 16            // W chunk = 16*128 rows * 16KB = 32 MB (L2-resident)

// pack two fp32 -> two bf16 (round-to-nearest via +0x8000 on magnitude bits),
// one v_add_nc_u32 each + one v_perm_b32 for the pair.
__device__ __forceinline__ unsigned int bfpack_round(float a, float b) {
  unsigned int ua = __builtin_bit_cast(unsigned int, a) + 0x8000u;
  unsigned int ub = __builtin_bit_cast(unsigned int, b) + 0x8000u;
  return __builtin_amdgcn_perm(ub, ua, 0x07060302u);  // {hi16(ub), hi16(ua)}
}

// pack two (q - zp) -> two bf16 by truncation: exact for int8-valued q, zp=0.
__device__ __forceinline__ unsigned int wpack(int qa, int qb, float z) {
  float fa = (float)qa - z;
  float fb = (float)qb - z;
  unsigned int ua = __builtin_bit_cast(unsigned int, fa);
  unsigned int ub = __builtin_bit_cast(unsigned int, fb);
  return __builtin_amdgcn_perm(ub, ua, 0x07060302u);
}

__global__ __launch_bounds__(256) void qlinear_bf16_wmma(
    const float* __restrict__ X,      // [M, K] fp32
    const int*   __restrict__ Wq,     // [N, K] int32 (int8-valued)
    const float* __restrict__ scale,  // scalar
    const float* __restrict__ zpoint, // scalar
    const float* __restrict__ bias,   // [N]
    float*       __restrict__ Y) {    // [M, N]
  __shared__ unsigned short sX[2][BM][LDS_STRIDE];
  __shared__ unsigned short sW[2][BN][LDS_STRIDE];

  const int t    = threadIdx.x;
  const int lane = t & 31;
  const int wave = t >> 5;      // 0..7
  const int wm   = wave >> 1;   // 0..3 (M direction, 32 rows each)
  const int wn   = wave & 1;    // 0..1 (N direction, 64 cols each)

  // L2-aware swizzle: 1024 blocks per chunk = 16 N-blocks x 64 M-blocks,
  // with the 16 same-M blocks adjacent in dispatch order.
  const int id   = blockIdx.x;
  const int c    = id >> 10;                 // chunk 0..7
  const int r    = id & 1023;
  const int m0   = (r >> 4) * BM;            // M block 0..63
  const int n0   = ((c << 4) | (r & 15)) * BN;

  const float s = scale[0];
  const float z = zpoint[0];

  // Staging map: 2 threads per row, 16 elements (64B global) each.
  const int sr = t >> 1;          // 0..127
  const int sc = (t & 1) * 16;    // 0 or 16
  const float* xg = X  + (size_t)(m0 + sr) * K_DIM + sc;
  const int*   wg = Wq + (size_t)(n0 + sr) * K_DIM + sc;

  float4 gx[4];
  int4   gw[4];
  {
    const float4* p = (const float4*)xg;
    const int4*   q = (const int4*)wg;
#pragma unroll
    for (int i = 0; i < 4; ++i) { gx[i] = p[i]; gw[i] = q[i]; }
  }

  v8f acc[2][4];
#pragma unroll
  for (int i = 0; i < 2; ++i)
#pragma unroll
    for (int j = 0; j < 4; ++j)
#pragma unroll
      for (int e = 0; e < 8; ++e) acc[i][j][e] = 0.0f;

  int buf = 0;
  for (int kb = 0; kb < NKB; ++kb) {
    // ---- issue next tile's global loads ----
    float4 nx[4];
    int4   nw[4];
    if (kb + 1 < NKB) {
      const float4* p = (const float4*)(xg + (size_t)(kb + 1) * BK);
      const int4*   q = (const int4*)(wg + (size_t)(kb + 1) * BK);
#pragma unroll
      for (int i = 0; i < 4; ++i) { nx[i] = p[i]; nw[i] = q[i]; }
    }
    if (kb + 2 < NKB) {  // warm L2 -> global_prefetch_b8
      __builtin_prefetch(xg + (size_t)(kb + 2) * BK, 0, 1);
      __builtin_prefetch(wg + (size_t)(kb + 2) * BK, 0, 1);
    }

    // ---- stage current regs -> LDS[buf] ----
    {
      uint4 px0, px1;
      px0.x = bfpack_round(gx[0].x, gx[0].y);
      px0.y = bfpack_round(gx[0].z, gx[0].w);
      px0.z = bfpack_round(gx[1].x, gx[1].y);
      px0.w = bfpack_round(gx[1].z, gx[1].w);
      px1.x = bfpack_round(gx[2].x, gx[2].y);
      px1.y = bfpack_round(gx[2].z, gx[2].w);
      px1.z = bfpack_round(gx[3].x, gx[3].y);
      px1.w = bfpack_round(gx[3].z, gx[3].w);
      *(uint4*)&sX[buf][sr][sc]     = px0;
      *(uint4*)&sX[buf][sr][sc + 8] = px1;

      uint4 pw0, pw1;
      pw0.x = wpack(gw[0].x, gw[0].y, z);
      pw0.y = wpack(gw[0].z, gw[0].w, z);
      pw0.z = wpack(gw[1].x, gw[1].y, z);
      pw0.w = wpack(gw[1].z, gw[1].w, z);
      pw1.x = wpack(gw[2].x, gw[2].y, z);
      pw1.y = wpack(gw[2].z, gw[2].w, z);
      pw1.z = wpack(gw[3].x, gw[3].y, z);
      pw1.w = wpack(gw[3].z, gw[3].w, z);
      *(uint4*)&sW[buf][sr][sc]     = pw0;
      *(uint4*)&sW[buf][sr][sc + 8] = pw1;
    }
    __syncthreads();

    // ---- fragments (lanes 0-15: K=0..15, lanes 16-31: K=16..31) ----
    const int ln = lane & 15;
    const int hk = (lane >> 4) * 16;
    v16bf afrag[2], bfrag[4];
#pragma unroll
    for (int tm = 0; tm < 2; ++tm) {
      const unsigned short* p = &sX[buf][wm * 32 + tm * 16 + ln][hk];
      union { uint4 q[2]; v16bf v; } u;
      u.q[0] = *(const uint4*)p;
      u.q[1] = *(const uint4*)(p + 8);
      afrag[tm] = u.v;
    }
#pragma unroll
    for (int tn = 0; tn < 4; ++tn) {
      const unsigned short* p = &sW[buf][wn * 64 + tn * 16 + ln][hk];
      union { uint4 q[2]; v16bf v; } u;
      u.q[0] = *(const uint4*)p;
      u.q[1] = *(const uint4*)(p + 8);
      bfrag[tn] = u.v;
    }

    // ---- 2x4 of 16x16x32 bf16 WMMA per wave ----
#pragma unroll
    for (int tm = 0; tm < 2; ++tm)
#pragma unroll
      for (int tn = 0; tn < 4; ++tn)
        acc[tm][tn] = __builtin_amdgcn_wmma_f32_16x16x32_bf16(
            false, afrag[tm], false, bfrag[tn], (short)0, acc[tm][tn],
            false, false);

    __syncthreads();
    if (kb + 1 < NKB) {
#pragma unroll
      for (int i = 0; i < 4; ++i) { gx[i] = nx[i]; gw[i] = nw[i]; }
    }
    buf ^= 1;
  }

  // ---- epilogue: Y = s * acc + bias ----
  const int ln  = lane & 15;
  const int mhi = (lane >> 4) * 8;
#pragma unroll
  for (int tn = 0; tn < 4; ++tn) {
    const int col = n0 + wn * 64 + tn * 16 + ln;
    const float bv = bias[col];
#pragma unroll
    for (int tm = 0; tm < 2; ++tm) {
      const int row0 = m0 + wm * 32 + tm * 16 + mhi;
#pragma unroll
      for (int j = 0; j < 8; ++j) {
        Y[(size_t)(row0 + j) * N_DIM + col] = fmaf(s, acc[tm][tn][j], bv);
      }
    }
  }
}

extern "C" void kernel_launch(void* const* d_in, const int* in_sizes, int n_in,
                              void* d_out, int out_size, void* d_ws, size_t ws_size,
                              hipStream_t stream) {
  const float* X      = (const float*)d_in[0];
  const int*   Wq     = (const int*)d_in[1];
  const float* scale  = (const float*)d_in[2];
  const float* zpoint = (const float*)d_in[3];
  const float* bias   = (const float*)d_in[4];
  float*       Y      = (float*)d_out;

  qlinear_bf16_wmma<<<dim3(MBLKS * NBLKS), 256, 0, stream>>>(
      X, Wq, scale, zpoint, bias, Y);
}
